// MultiHeadLatentAttention_46342697123961
// MI455X (gfx1250) — matrix operand order repaired and checked by reading, not compiled, not run on hardware
//
#include <hip/hip_runtime.h>

// Problem constants (match reference)
#define BB     4
#define TT     2048
#define DD     1024
#define HH     16
#define HDD    64
#define LATD   128
#define ROPED  32
#define BT     (BB * TT)        // 8192
#define KLD    (HH * ROPED)     // 512, K row stride
#define VLD    (HH * 32)        // 512, V row stride

typedef __attribute__((ext_vector_type(16))) __bf16       v16bf;
typedef __attribute__((ext_vector_type(8)))  float        v8f;
typedef __attribute__((ext_vector_type(8)))  unsigned int v8u;
typedef unsigned short ushort_t;

// round-to-nearest-even fp32 -> bf16 helpers (used by the prep passes and the attention kernel)
__device__ __forceinline__ unsigned int pk_bf16(float lo, float hi) {
  unsigned int a = __builtin_bit_cast(unsigned int, lo);
  unsigned int b = __builtin_bit_cast(unsigned int, hi);
  a = (a + 0x7FFFu + ((a >> 16) & 1u)) >> 16;
  b = (b + 0x7FFFu + ((b >> 16) & 1u)) >> 16;
  return (a & 0xFFFFu) | (b << 16);
}
__device__ __forceinline__ ushort_t bf16_1(float f) {
  unsigned int u = __builtin_bit_cast(unsigned int, f);
  return (ushort_t)((u + 0x7FFFu + ((u >> 16) & 1u)) >> 16);
}

__device__ __forceinline__ v16bf frag_from(uint4 lo, uint4 hi) {
  v8u u;
  u[0] = lo.x; u[1] = lo.y; u[2] = lo.z; u[3] = lo.w;
  u[4] = hi.x; u[5] = hi.y; u[6] = hi.z; u[7] = hi.w;
  return __builtin_bit_cast(v16bf, u);
}

// ---------------------------------------------------------------------------
// One-shot fp32 -> bf16 flat conversion (4 elem / thread, 64-bit stores)
// ---------------------------------------------------------------------------
__global__ __launch_bounds__(256)
void cvt_f32_bf16(const float* __restrict__ X, ushort_t* __restrict__ Y, long n)
{
  long i = ((long)blockIdx.x * blockDim.x + threadIdx.x) * 4;
  if (i >= n) return;
  float4 v = *(const float4*)(X + i);
  uint2 o;
  o.x = pk_bf16(v.x, v.y);
  o.y = pk_bf16(v.z, v.w);
  *(uint2*)(Y + i) = o;
}

// ---------------------------------------------------------------------------
// One-shot weight transpose + convert: W[K][N] fp32 -> Wt[N][K] bf16.
// LDS-tiled 32x32 so both global accesses are coalesced.
// ---------------------------------------------------------------------------
__global__ __launch_bounds__(256)
void transpose_cvt_bf16(const float* __restrict__ W, ushort_t* __restrict__ Wt,
                        int K, int N)
{
  __shared__ ushort_t tile[32][33];
  const int n0 = blockIdx.x * 32, k0 = blockIdx.y * 32;
  const int tx = threadIdx.x, ty = threadIdx.y;       // 32 x 8
#pragma unroll
  for (int i = 0; i < 32; i += 8)
    tile[ty + i][tx] = bf16_1(W[(size_t)(k0 + ty + i) * N + n0 + tx]);
  __syncthreads();
#pragma unroll
  for (int i = 0; i < 32; i += 8)
    Wt[(size_t)(n0 + ty + i) * K + k0 + tx] = tile[tx][ty + i];
}

// ---------------------------------------------------------------------------
// bf16 WMMA GEMM: C[M,N](fp32) = A[M,K](bf16,row-major) * Wt[N,K](bf16).
// One wave computes a 16x64 tile (one A fragment reused across 4 B frags).
// Ping-pong software pipeline via manual 2x unroll: set1 (k+32) loads are in
// flight while set0 (k) feeds 4 v_wmma ops, then set0 reloads in place at
// k+64 while set1 is consumed -- no register rotation, no v_mov chains.
// KC is compile-time (1024 or 128), a multiple of 64.
// ---------------------------------------------------------------------------
template <int KC>
__global__ __launch_bounds__(256)
void gemm_bf16t_wmma(const ushort_t* __restrict__ A, const ushort_t* __restrict__ Bt,
                     float* __restrict__ C, int M, int N)
{
  const int lane = threadIdx.x & 31;
  const int wid  = threadIdx.x >> 5;
  const int tn   = N >> 6;                      // 64-wide N tiles
  const int t    = blockIdx.x * (blockDim.x >> 5) + wid;
  if (t >= (M >> 4) * tn) return;
  const int m0   = (t / tn) << 4;
  const int n0   = (t % tn) << 6;
  const int r    = lane & 15;
  const int half = lane >> 4;

  const ushort_t* arow = A + (size_t)(m0 + r) * KC + half * 16;
  const ushort_t* brow[4];
#pragma unroll
  for (int s = 0; s < 4; ++s)
    brow[s] = Bt + (size_t)(n0 + s * 16 + r) * KC + half * 16;

  // set0: fragments for k = 0
  uint4 a0_lo = *(const uint4*)(arow);
  uint4 a0_hi = *(const uint4*)(arow + 8);
  uint4 b0_lo[4], b0_hi[4];
#pragma unroll
  for (int s = 0; s < 4; ++s) {
    b0_lo[s] = *(const uint4*)(brow[s]);
    b0_hi[s] = *(const uint4*)(brow[s] + 8);
  }

  v8f acc[4] = {{}, {}, {}, {}};
  for (int k = 0; k < KC; k += 64) {
    // set1 loads (k+32) in flight while set0 is consumed
    const uint4 a1_lo = *(const uint4*)(arow + k + 32);
    const uint4 a1_hi = *(const uint4*)(arow + k + 40);
    uint4 b1_lo[4], b1_hi[4];
#pragma unroll
    for (int s = 0; s < 4; ++s) {
      b1_lo[s] = *(const uint4*)(brow[s] + k + 32);
      b1_hi[s] = *(const uint4*)(brow[s] + k + 40);
    }
    if (k + 64 < KC) __builtin_prefetch(arow + k + 64, 0, 1);

    const v16bf a0 = frag_from(a0_lo, a0_hi);
#pragma unroll
    for (int s = 0; s < 4; ++s)
      acc[s] = __builtin_amdgcn_wmma_f32_16x16x32_bf16(
          false, a0, false, frag_from(b0_lo[s], b0_hi[s]), (short)0, acc[s], false, false);

    // set0 reloads in place (k+64) while set1 is consumed
    if (k + 64 < KC) {
      a0_lo = *(const uint4*)(arow + k + 64);
      a0_hi = *(const uint4*)(arow + k + 72);
#pragma unroll
      for (int s = 0; s < 4; ++s) {
        b0_lo[s] = *(const uint4*)(brow[s] + k + 64);
        b0_hi[s] = *(const uint4*)(brow[s] + k + 72);
      }
    }

    const v16bf a1 = frag_from(a1_lo, a1_hi);
#pragma unroll
    for (int s = 0; s < 4; ++s)
      acc[s] = __builtin_amdgcn_wmma_f32_16x16x32_bf16(
          false, a1, false, frag_from(b1_lo[s], b1_hi[s]), (short)0, acc[s], false, false);
  }

  float* crow = C + (size_t)(m0 + half * 8) * N + n0 + r;
#pragma unroll
  for (int s = 0; s < 4; ++s)
#pragma unroll
    for (int i = 0; i < 8; ++i)
      crow[(size_t)i * N + s * 16] = acc[s][i];
}

// ---------------------------------------------------------------------------
// RMSNorm over contiguous groups of G (=64 or 32), fused fp32 -> bf16 output.
// One wave per group; full-wave shuffle reduction.
// ---------------------------------------------------------------------------
__global__ __launch_bounds__(256)
void rmsnorm_cvt(const float* __restrict__ X, const float* __restrict__ g,
                 ushort_t* __restrict__ Y, int groups, int G)
{
  const int gid  = (blockIdx.x * blockDim.x + threadIdx.x) >> 5;
  const int lane = threadIdx.x & 31;
  if (gid >= groups) return;
  const float* x = X + (size_t)gid * G;
  ushort_t*    y = Y + (size_t)gid * G;

  float v0 = x[lane];
  float v1 = 0.f;
  if (G == 64) v1 = x[lane + 32];
  float ss = v0 * v0 + v1 * v1;
#pragma unroll
  for (int m = 16; m >= 1; m >>= 1) ss += __shfl_xor(ss, m, 32);
  const float s = rsqrtf(ss / (float)G + 1e-6f);
  y[lane] = bf16_1(v0 * s * g[lane]);
  if (G == 64) y[lane + 32] = bf16_1(v1 * s * g[lane + 32]);
}

// ---------------------------------------------------------------------------
// V relayout: V[(b*T+key)][h*32+dim] fp32 -> Vt[((b*H+h)*32+dim)][key] bf16.
// Key-contiguous so the P*V B-fragment is two b128 loads per lane.
// ---------------------------------------------------------------------------
__global__ __launch_bounds__(256)
void v_transpose_cvt(const float* __restrict__ V, ushort_t* __restrict__ Vt)
{
  __shared__ ushort_t tile[32][33];
  const int k0 = blockIdx.x * 32;     // key tile
  const int h  = blockIdx.y;
  const int b  = blockIdx.z;
  const int tx = threadIdx.x, ty = threadIdx.y;  // 32 x 8
#pragma unroll
  for (int i = 0; i < 32; i += 8)     // tile[key][dim]
    tile[ty + i][tx] = bf16_1(V[(size_t)(b * TT + k0 + ty + i) * VLD + h * 32 + tx]);
  __syncthreads();
#pragma unroll
  for (int i = 0; i < 32; i += 8)
    Vt[((size_t)(b * HH + h) * 32 + ty + i) * TT + k0 + tx] = tile[tx][ty + i];
}

// ---------------------------------------------------------------------------
// Flash attention, one wave per 16-query tile per (b,h), all operands bf16.
// Software-pipelined: after the S = Q*K^T WMMAs issue, the V fragments for
// this chunk and the K fragments for the next chunk (written in place, no
// rotation) are loaded while the softmax VALU runs; then P is restaged via
// LDS into the A layout and O += P*V (2 WMMAs). Emits bf16 output incl.
// the zero-padded dims 32..63.
// ---------------------------------------------------------------------------
__global__ __launch_bounds__(32)
void mla_flash_attn(const ushort_t* __restrict__ Qb, const ushort_t* __restrict__ Kb,
                    const ushort_t* __restrict__ Vt, ushort_t* __restrict__ AO)
{
  __shared__ ushort_t p_lds[16 * 32];   // P tile, bf16
  const int lane = threadIdx.x;
  const int n    = lane & 15;
  const int half = lane >> 4;
  const int t0   = blockIdx.x << 4;
  const int h    = blockIdx.y;
  const int b    = blockIdx.z;

  // Q fragment (A layout): rows t0..t0+15 of head h, rope dims 0..31
  const ushort_t* qp = Qb + (size_t)(b * TT + t0 + n) * DD + h * HDD + half * 16;
  const v16bf aq = frag_from(*(const uint4*)qp, *(const uint4*)(qp + 8));

  // per-chunk base pointers (chunk kc: kbase + kc*KLD, vbase + kc)
  const ushort_t* kbase = Kb + (size_t)(b * TT + n) * KLD + h * ROPED + half * 16;
  const ushort_t* vbase = Vt + ((size_t)(b * HH + h) * 32 + n) * TT + half * 16;

  v8f o0 = {}, o1 = {};
  float mrow[8], lrow[8];
#pragma unroll
  for (int i = 0; i < 8; ++i) { mrow[i] = -__builtin_inff(); lrow[i] = 0.f; }

  const float scale = 0.125f;   // 1/sqrt(HD=64)
  const int   kend  = t0 + 16;

  // prologue: K fragments for chunk 0
  uint4 k0_lo = *(const uint4*)(kbase);
  uint4 k0_hi = *(const uint4*)(kbase + 8);
  uint4 k1_lo = *(const uint4*)(kbase + (size_t)16 * KLD);
  uint4 k1_hi = *(const uint4*)(kbase + (size_t)16 * KLD + 8);

  for (int kc = 0; kc < kend; kc += 32) {
    v8f s0 = {}, s1 = {};
    s0 = __builtin_amdgcn_wmma_f32_16x16x32_bf16(false, aq, false,
             frag_from(k0_lo, k0_hi), (short)0, s0, false, false);
    s1 = __builtin_amdgcn_wmma_f32_16x16x32_bf16(false, aq, false,
             frag_from(k1_lo, k1_hi), (short)0, s1, false, false);

    // V fragments for this chunk (independent of softmax -> issue now)
    const ushort_t* vp = vbase + kc;
    const uint4 v0_lo = *(const uint4*)(vp);
    const uint4 v0_hi = *(const uint4*)(vp + 8);
    const uint4 v1_lo = *(const uint4*)(vp + (size_t)16 * TT);
    const uint4 v1_hi = *(const uint4*)(vp + (size_t)16 * TT + 8);

    // next chunk's K fragments in flight under the softmax VALU (in-place)
    if (kc + 32 < kend) {
      const ushort_t* kp = kbase + (size_t)(kc + 32) * KLD;
      k0_lo = *(const uint4*)(kp);
      k0_hi = *(const uint4*)(kp + 8);
      k1_lo = *(const uint4*)(kp + (size_t)16 * KLD);
      k1_hi = *(const uint4*)(kp + (size_t)16 * KLD + 8);
    }

    // scale, causal mask, online softmax (row stats replicated per 16-lane half)
    float p0[8], p1[8], alpha[8];
#pragma unroll
    for (int i = 0; i < 8; ++i) {
      const int row = t0 + i + 8 * half;
      float a = s0[i] * scale;
      float c = s1[i] * scale;
      if (kc + n > row)      a = -__builtin_inff();
      if (kc + 16 + n > row) c = -__builtin_inff();
      float cm = fmaxf(a, c);
      cm = fmaxf(cm, __shfl_xor(cm, 1, 32));
      cm = fmaxf(cm, __shfl_xor(cm, 2, 32));
      cm = fmaxf(cm, __shfl_xor(cm, 4, 32));
      cm = fmaxf(cm, __shfl_xor(cm, 8, 32));
      const float mnew = fmaxf(mrow[i], cm);
      const float al   = __expf(mrow[i] - mnew);
      const float e0   = __expf(a - mnew);
      const float e1   = __expf(c - mnew);
      float rs = e0 + e1;
      rs += __shfl_xor(rs, 1, 32);
      rs += __shfl_xor(rs, 2, 32);
      rs += __shfl_xor(rs, 4, 32);
      rs += __shfl_xor(rs, 8, 32);
      lrow[i]  = lrow[i] * al + rs;
      mrow[i]  = mnew;
      alpha[i] = al;
      p0[i] = e0; p1[i] = e1;
    }
#pragma unroll
    for (int i = 0; i < 8; ++i) { o0[i] *= alpha[i]; o1[i] *= alpha[i]; }

    // restage P (C layout) -> LDS -> A layout for the PV WMMA
#pragma unroll
    for (int i = 0; i < 8; ++i) {
      const int rl = i + 8 * half;
      p_lds[rl * 32 + n]      = bf16_1(p0[i]);
      p_lds[rl * 32 + 16 + n] = bf16_1(p1[i]);
    }
    __syncthreads();
    v8u pau;
    {
      const unsigned int* p32 = (const unsigned int*)p_lds;
#pragma unroll
      for (int j = 0; j < 4; ++j) {
        pau[j]     = p32[n * 16 + half * 8 + j];
        pau[4 + j] = p32[n * 16 + half * 8 + 4 + j];
      }
    }
    __syncthreads();
    const v16bf pa = __builtin_bit_cast(v16bf, pau);

    o0 = __builtin_amdgcn_wmma_f32_16x16x32_bf16(false, pa, false,
             frag_from(v0_lo, v0_hi), (short)0, o0, false, false);
    o1 = __builtin_amdgcn_wmma_f32_16x16x32_bf16(false, pa, false,
             frag_from(v1_lo, v1_hi), (short)0, o1, false, false);
  }

  // epilogue: normalize by row sums, emit bf16 head slice (dims 32..63 = 0 pad)
#pragma unroll
  for (int i = 0; i < 8; ++i) {
    const int t = t0 + i + 8 * half;
    const float inv = 1.0f / lrow[i];
    ushort_t* op = AO + (size_t)(b * TT + t) * DD + h * HDD;
    op[n]      = bf16_1(o0[i] * inv);
    op[16 + n] = bf16_1(o1[i] * inv);
    op[32 + n] = 0;
    op[48 + n] = 0;
  }
}

// ---------------------------------------------------------------------------
extern "C" void kernel_launch(void* const* d_in, const int* in_sizes, int n_in,
                              void* d_out, int out_size, void* d_ws, size_t ws_size,
                              hipStream_t stream)
{
  (void)in_sizes; (void)n_in; (void)out_size; (void)ws_size;
  const float* x   = (const float*)d_in[0];
  const float* Wq  = (const float*)d_in[1];
  const float* Wkv = (const float*)d_in[2];
  const float* Wk  = (const float*)d_in[3];
  const float* Wv  = (const float*)d_in[4];
  const float* Wo  = (const float*)d_in[5];
  const float* qg  = (const float*)d_in[6];
  const float* kg  = (const float*)d_in[7];
  float* out = (float*)d_out;

  // workspace carve-out (all 256B aligned)
  char* w = (char*)d_ws;
  auto carve = [&](size_t bytes) { void* p = (void*)w; w += (bytes + 255) & ~(size_t)255; return p; };
  float*    Qf   = (float*)   carve((size_t)BT * DD * 4);     // Q proj fp32
  float*    KVf  = (float*)   carve((size_t)BT * LATD * 4);   // latent fp32
  float*    Kf   = (float*)   carve((size_t)BT * KLD * 4);    // K proj fp32
  float*    Vf   = (float*)   carve((size_t)BT * VLD * 4);    // V proj fp32
  ushort_t* xbf  = (ushort_t*)carve((size_t)BT * DD * 2);
  ushort_t* KVbf = (ushort_t*)carve((size_t)BT * LATD * 2);
  ushort_t* Qbf  = (ushort_t*)carve((size_t)BT * DD * 2);     // normalized Q
  ushort_t* Kbf  = (ushort_t*)carve((size_t)BT * KLD * 2);    // normalized K
  ushort_t* Vtb  = (ushort_t*)carve((size_t)BT * VLD * 2);    // V, (b,h,dim,key)
  ushort_t* AObf = (ushort_t*)carve((size_t)BT * DD * 2);     // attn out
  ushort_t* WqT  = (ushort_t*)carve((size_t)DD * DD * 2);
  ushort_t* WkvT = (ushort_t*)carve((size_t)DD * LATD * 2);
  ushort_t* WkT  = (ushort_t*)carve((size_t)LATD * KLD * 2);
  ushort_t* WvT  = (ushort_t*)carve((size_t)LATD * VLD * 2);
  ushort_t* WoT  = (ushort_t*)carve((size_t)DD * DD * 2);

  auto cvt = [&](const float* X, ushort_t* Y, long n) {
    cvt_f32_bf16<<<dim3((unsigned)((n / 4 + 255) / 256)), 256, 0, stream>>>(X, Y, n);
  };
  auto wtrans = [&](const float* W, ushort_t* Wt, int K, int N) {
    transpose_cvt_bf16<<<dim3(N / 32, K / 32), dim3(32, 8), 0, stream>>>(W, Wt, K, N);
  };
  auto gemm = [&](const ushort_t* A, const ushort_t* Bt, float* C, int M, int N, int K) {
    const int waves  = (M / 16) * (N / 64);
    const dim3 grid((waves + 7) / 8);
    if (K == 1024)
      gemm_bf16t_wmma<1024><<<grid, 256, 0, stream>>>(A, Bt, C, M, N);
    else
      gemm_bf16t_wmma<128><<<grid, 256, 0, stream>>>(A, Bt, C, M, N);
  };

  // one-shot operand prep (bandwidth-trivial, stays resident in 192MB L2)
  cvt(x, xbf, (long)BT * DD);
  wtrans(Wq,  WqT,  DD,   DD);
  wtrans(Wkv, WkvT, DD,   LATD);
  wtrans(Wk,  WkT,  LATD, KLD);
  wtrans(Wv,  WvT,  LATD, VLD);
  wtrans(Wo,  WoT,  DD,   DD);

  // projections
  gemm(xbf, WqT,  Qf,  BT, DD,   DD);     // Q
  gemm(xbf, WkvT, KVf, BT, LATD, DD);     // latent KV
  cvt(KVf, KVbf, (long)BT * LATD);
  gemm(KVbf, WkT, Kf, BT, KLD, LATD);     // K rope
  gemm(KVbf, WvT, Vf, BT, VLD, LATD);     // V

  // normalize + relayout into attention-friendly bf16
  {
    const int groups = BT * HH;           // 131072
    rmsnorm_cvt<<<dim3((groups * 32 + 255) / 256), 256, 0, stream>>>(Qf, qg, Qbf, groups, HDD);
    rmsnorm_cvt<<<dim3((groups * 32 + 255) / 256), 256, 0, stream>>>(Kf, kg, Kbf, groups, ROPED);
    v_transpose_cvt<<<dim3(TT / 32, HH, BB), dim3(32, 8), 0, stream>>>(Vf, Vtb);
  }

  // causal latent attention
  mla_flash_attn<<<dim3(TT / 16, HH, BB), 32, 0, stream>>>(Qbf, Kbf, Vtb, AObf);

  // output projection -> fp32 result
  gemm(AObf, WoT, out, BT, DD, DD);
}
